// EMAVectorQuantizer_8976481649065
// MI455X (gfx1250) — matrix operand order repaired and checked by reading, not compiled, not run on hardware
//
#include <hip/hip_runtime.h>
#include <hip/hip_bf16.h>

// ---------------------------------------------------------------------------
// EMA Vector Quantizer for MI455X (gfx1250, wave32, WMMA)
// B=16, D=512, H=W=32 -> N=16384 vectors; K=2048 codes.
// Memory-bound on the 128MB distance_prob output; GEMM done with
// v_wmma_f32_16x16x32_f16, distance tile kept in 128KB of LDS per WG
// (CDNA5 has 320KB LDS/WGP) and softmax/argmin fused so distances never
// touch HBM.
// ---------------------------------------------------------------------------

typedef _Float16 f16;
typedef __attribute__((ext_vector_type(16))) _Float16 v16h;
typedef __attribute__((ext_vector_type(8)))  float    v8f;

constexpr int Bb   = 16;
constexpr int Dd   = 512;
constexpr int HW   = 32 * 32;        // 1024
constexpr int Nn   = Bb * HW;        // 16384
constexpr int Kk   = 2048;
constexpr float DECAY = 0.99f;
constexpr float ONEMD = 0.01f;
constexpr float EPSv  = 1e-5f;
constexpr float BETA  = 0.25f;

// ---------------------------------------------------------------- utilities
__global__ void zero_kernel(float* __restrict__ p, long n) {
  long i = (long)blockIdx.x * blockDim.x + threadIdx.x;
  if (i < n) p[i] = 0.0f;
}

// ------------------------------------------------- codebook normalize -> f16
__global__ void __launch_bounds__(256)
cbnorm_kernel(const float* __restrict__ emb, f16* __restrict__ Ch,
              float* __restrict__ csqn) {
  __shared__ float red[256];
  const int k = blockIdx.x;
  const float* row = emb + (long)k * Dd;
  float ss = 0.f;
  for (int d = threadIdx.x; d < Dd; d += 256) { float v = row[d]; ss += v * v; }
  red[threadIdx.x] = ss; __syncthreads();
  for (int s = 128; s > 0; s >>= 1) {
    if (threadIdx.x < s) red[threadIdx.x] += red[threadIdx.x + s];
    __syncthreads();
  }
  const float sumsq = red[0];
  const float invn  = 1.0f / fmaxf(sqrtf(sumsq), 1e-12f);
  if (threadIdx.x == 0) csqn[k] = sumsq * invn * invn; // sum(cb_norm^2)
  for (int d = threadIdx.x; d < Dd; d += 256)
    Ch[(long)k * Dd + d] = (f16)(row[d] * invn);
}

// ---------------------------------- per-row sumsq of z (gather over D axis)
__global__ void __launch_bounds__(256)
zsumsq_kernel(const float* __restrict__ z, float* __restrict__ inv,
              float* __restrict__ zsqn) {
  __shared__ float red[256];
  const int n  = blockIdx.x;
  const int b  = n >> 10;        // HW = 1024
  const int hw = n & 1023;
  float ss = 0.f;
  for (int d = threadIdx.x; d < Dd; d += 256) {
    float v = z[(((long)(b * Dd + d)) << 10) + hw];
    ss += v * v;
  }
  red[threadIdx.x] = ss; __syncthreads();
  for (int s = 128; s > 0; s >>= 1) {
    if (threadIdx.x < s) red[threadIdx.x] += red[threadIdx.x + s];
    __syncthreads();
  }
  if (threadIdx.x == 0) {
    float sumsq = red[0];
    float iv = 1.0f / fmaxf(sqrtf(sumsq), 1e-12f);
    inv[n]  = iv;
    zsqn[n] = sumsq * iv * iv;   // sum(z_norm^2)
  }
}

// --------------- [B,D,H,W] -> [N,D] transpose + normalize, f32 + f16 copies
__global__ void __launch_bounds__(1024)
tnorm_kernel(const float* __restrict__ z, const float* __restrict__ inv,
             float* __restrict__ zn, f16* __restrict__ Zh) {
  __shared__ float tile[32][33];
  const int b   = blockIdx.z;
  const int d0  = blockIdx.y * 32;
  const int hw0 = blockIdx.x * 32;
  const int tx = threadIdx.x, ty = threadIdx.y;
  // coalesced read along hw
  tile[ty][tx] = z[(((long)(b * Dd + d0 + ty)) << 10) + hw0 + tx];
  __syncthreads();
  // coalesced write along d
  const int n = b * HW + hw0 + ty;
  const int d = d0 + tx;
  const float v = tile[tx][ty] * inv[n];
  zn[(long)n * Dd + d] = v;
  Zh[(long)n * Dd + d] = (f16)v;
}

// ---------------------------------------------------------------------------
// Core: 16-row stripe per workgroup. 8 waves each own 16 column tiles of 16.
// WMMA f32_16x16x32_f16 over D=512 (16 k-chunks). Distances -> 128KB LDS,
// then fused softmax(-d) + argmin, probs streamed once to HBM.
// ---------------------------------------------------------------------------
__global__ void __launch_bounds__(256)
gemm_softmax_kernel(const f16* __restrict__ Zh, const f16* __restrict__ Ch,
                    const float* __restrict__ zsqn, const float* __restrict__ csqn,
                    float* __restrict__ prob, int* __restrict__ indices) {
  extern __shared__ float dist[];              // [16][Kk] = 128 KB
  const int r0   = blockIdx.x * 16;
  const int wave = threadIdx.x >> 5;
  const int lane = threadIdx.x & 31;
  const int m16  = lane & 15;
  const int hi   = lane >> 4;

  const long arow = (long)(r0 + m16) * Dd;     // A: row r0+m16 of Zh

  for (int tile = wave; tile < Kk / 16; tile += 8) {
    const int c0 = tile * 16;
    const long brow = (long)(c0 + m16) * Dd;   // B column n = codebook row c0+n
    v8f acc = {};
    for (int kc = 0; kc < Dd; kc += 32) {
      v16h a, b;
#pragma unroll
      for (int e = 0; e < 16; ++e) {
        const int g = e >> 1, pos = e & 1;
        // A 16x32 f16 lane layout (ISA 7.12.2)
        const int ka = ((g < 4) ? (2 * g) : (2 * g + 8)) + 8 * hi + pos;
        // B 32x16 f16 lane layout
        const int kb = ((g < 4) ? (2 * g) : (2 * g + 8)) + 8 * hi + pos;
        a[e] = Zh[arow + kc + ka];
        b[e] = Ch[brow + kc + kb];
      }
      acc = __builtin_amdgcn_wmma_f32_16x16x32_f16(
          false, a, false, b, (short)0, acc, false, false);
    }
    // D layout: vgpr v -> m = 8*hi + v, n = lane&15
#pragma unroll
    for (int v = 0; v < 8; ++v) {
      const int m = 8 * hi + v;
      const int n = m16;
      const float dval = zsqn[r0 + m] + csqn[c0 + n] - 2.0f * acc[v];
      dist[m * Kk + (c0 + n)] = dval;
    }
  }
  __syncthreads();

  // softmax + argmin: each wave owns 2 rows
  for (int rr = 0; rr < 2; ++rr) {
    const int r = wave * 2 + rr;
    float mn = 3.4e38f; int mi = 0;
    for (int j = lane; j < Kk; j += 32) {
      const float dv = dist[r * Kk + j];
      if (dv < mn) { mn = dv; mi = j; }
    }
#pragma unroll
    for (int off = 16; off >= 1; off >>= 1) {
      const float omn = __shfl_xor(mn, off, 32);
      const int   omi = __shfl_xor(mi, off, 32);
      if (omn < mn || (omn == mn && omi < mi)) { mn = omn; mi = omi; }
    }
    float s = 0.f;
    for (int j = lane; j < Kk; j += 32)
      s += __expf(mn - dist[r * Kk + j]);
#pragma unroll
    for (int off = 16; off >= 1; off >>= 1) s += __shfl_xor(s, off, 32);
    const float isum = 1.0f / s;
    const long rowbase = (long)(r0 + r) * Kk;
    for (int j = lane; j < Kk; j += 32)        // coalesced 128B stores
      prob[rowbase + j] = __expf(mn - dist[r * Kk + j]) * isum;
    if (lane == 0) indices[r0 + r] = mi;
  }
}

// ------------------------------------------ segment sums (n_total/encode_sum)
__global__ void __launch_bounds__(256)
stats_kernel(const float* __restrict__ zn, const int* __restrict__ idx,
             float* __restrict__ n_total, float* __restrict__ enc_sum) {
  const int n = blockIdx.x;
  const int k = idx[n];
  for (int d = threadIdx.x; d < Dd; d += 256)
    atomicAdd(&enc_sum[(long)k * Dd + d], zn[(long)n * Dd + d]);
  if (threadIdx.x == 0) atomicAdd(&n_total[k], 1.0f);
}

// -------------------------------------------------------------- EMA updates
__global__ void Nnew_kernel(const float* __restrict__ N_buf,
                            const float* __restrict__ n_total,
                            float* __restrict__ N_new) {
  const int k = blockIdx.x * blockDim.x + threadIdx.x;
  if (k < Kk) N_new[k] = DECAY * N_buf[k] + ONEMD * n_total[k];
}

__global__ void __launch_bounds__(256)
nsum_kernel(const float* __restrict__ N_new, float* __restrict__ nsum) {
  __shared__ float red[256];
  float s = 0.f;
  for (int k = threadIdx.x; k < Kk; k += 256) s += N_new[k];
  red[threadIdx.x] = s; __syncthreads();
  for (int st = 128; st > 0; st >>= 1) {
    if (threadIdx.x < st) red[threadIdx.x] += red[threadIdx.x + st];
    __syncthreads();
  }
  if (threadIdx.x == 0) nsum[0] = red[0];
}

__global__ void __launch_bounds__(256)
embnew_kernel(const float* __restrict__ z_avg, const float* __restrict__ enc_sum,
              const float* __restrict__ N_new, const float* __restrict__ nsum_p,
              float* __restrict__ z_avg_out, float* __restrict__ emb_out,
              float* __restrict__ cbsum_out) {
  __shared__ float red[256];
  const int k = blockIdx.x;
  const float n = nsum_p[0];
  const float w = (N_new[k] + EPSv) / (n + (float)Kk * EPSv) * n;
  const float invw = 1.0f / w;
  float la = 0.f;
  for (int d = threadIdx.x; d < Dd; d += 256) {
    const long i = (long)k * Dd + d;
    const float za = DECAY * z_avg[i] + ONEMD * enc_sum[i];
    z_avg_out[i] = za;
    const float e = za * invw;
    emb_out[i] = e;
    la += fabsf(e);
  }
  red[threadIdx.x] = la; __syncthreads();
  for (int s = 128; s > 0; s >>= 1) {
    if (threadIdx.x < s) red[threadIdx.x] += red[threadIdx.x + s];
    __syncthreads();
  }
  if (threadIdx.x == 0) atomicAdd(cbsum_out, red[0]);
}

// ------------------------ gather emb = z_norm[idx] + commitment loss partial
__global__ void __launch_bounds__(256)
gather_loss_kernel(const float* __restrict__ zn, const int* __restrict__ idx,
                   float* __restrict__ embbuf, float* __restrict__ loss_acc) {
  __shared__ float red[256];
  const int n = blockIdx.x;
  const int k = idx[n];
  float la = 0.f;
  for (int d = threadIdx.x; d < Dd; d += 256) {
    const float e  = zn[(long)k * Dd + d];
    embbuf[(long)n * Dd + d] = e;
    const float df = zn[(long)n * Dd + d] - e;
    la += df * df;
  }
  red[threadIdx.x] = la; __syncthreads();
  for (int s = 128; s > 0; s >>= 1) {
    if (threadIdx.x < s) red[threadIdx.x] += red[threadIdx.x + s];
    __syncthreads();
  }
  if (threadIdx.x == 0) atomicAdd(loss_acc, red[0]);
}

__global__ void finalize_kernel(const float* __restrict__ loss_acc,
                                float* __restrict__ commit_out,
                                float* __restrict__ loss_out) {
  const float c = loss_acc[0] / (float)((long)Nn * Dd);
  commit_out[0] = c;
  loss_out[0]   = BETA * c;
}

// --------------------------------------- [N,D] emb -> q [B,D,H,W] transpose
__global__ void __launch_bounds__(1024)
qtranspose_kernel(const float* __restrict__ embbuf, float* __restrict__ q) {
  __shared__ float tile[32][33];
  const int b   = blockIdx.z;
  const int d0  = blockIdx.y * 32;
  const int hw0 = blockIdx.x * 32;
  const int tx = threadIdx.x, ty = threadIdx.y;
  // coalesced read along d
  tile[ty][tx] = embbuf[(long)(b * HW + hw0 + ty) * Dd + d0 + tx];
  __syncthreads();
  // coalesced write along hw
  q[(((long)(b * Dd + d0 + ty)) << 10) + hw0 + tx] = tile[tx][ty];
}

// ---------------------------------------------------------------------------
extern "C" void kernel_launch(void* const* d_in, const int* in_sizes, int n_in,
                              void* d_out, int out_size, void* d_ws, size_t ws_size,
                              hipStream_t stream) {
  const float* z          = (const float*)d_in[0];  // [B,D,H,W]
  const float* embeddings = (const float*)d_in[1];  // [K,D]
  const float* N_buf      = (const float*)d_in[2];  // [K]
  const float* z_avg      = (const float*)d_in[3];  // [K,D]

  float* out = (float*)d_out;
  float* q_out       = out;                          // N*D
  float* prob_out    = q_out + (long)Nn * Dd;        // N*K
  float* commit_out  = prob_out + (long)Nn * Kk;     // 1
  float* loss_out    = commit_out + 1;               // 1
  float* cbsum_out   = commit_out + 2;               // 1
  float* embnew_out  = commit_out + 3;               // K*D
  float* Nnew_out    = embnew_out + (long)Kk * Dd;   // K
  float* zavgnew_out = Nnew_out + Kk;                // K*D

  char* wbase = (char*)d_ws;
  size_t off = 0;
  auto take = [&](size_t bytes) -> char* {
    char* p = wbase + off;
    off = (off + bytes + 255) & ~(size_t)255;
    return p;
  };
  float* inv      = (float*)take((size_t)Nn * 4);
  float* zsqn     = (float*)take((size_t)Nn * 4);
  float* csqn     = (float*)take((size_t)Kk * 4);
  f16*   Zh       = (f16*)  take((size_t)Nn * Dd * 2);
  f16*   Ch       = (f16*)  take((size_t)Kk * Dd * 2);
  float* zn       = (float*)take((size_t)Nn * Dd * 4);
  float* embbuf   = (float*)take((size_t)Nn * Dd * 4);
  int*   indices  = (int*)  take((size_t)Nn * 4);
  float* n_total  = (float*)take((size_t)Kk * 4);
  float* enc_sum  = (float*)take((size_t)Kk * Dd * 4);
  float* nsum     = (float*)take(256);
  float* loss_acc = (float*)take(256);

  // zero accumulators
  zero_kernel<<<((long)Kk * Dd + 255) / 256, 256, 0, stream>>>(enc_sum, (long)Kk * Dd);
  zero_kernel<<<(Kk + 255) / 256, 256, 0, stream>>>(n_total, Kk);
  zero_kernel<<<1, 32, 0, stream>>>(loss_acc, 1);
  zero_kernel<<<1, 32, 0, stream>>>(cbsum_out, 1);

  // normalize inputs
  cbnorm_kernel<<<Kk, 256, 0, stream>>>(embeddings, Ch, csqn);
  zsumsq_kernel<<<Nn, 256, 0, stream>>>(z, inv, zsqn);
  tnorm_kernel<<<dim3(HW / 32, Dd / 32, Bb), dim3(32, 32), 0, stream>>>(z, inv, zn, Zh);

  // WMMA distance GEMM + fused softmax/argmin (128KB dynamic LDS per WG)
  gemm_softmax_kernel<<<Nn / 16, 256, 16 * Kk * sizeof(float), stream>>>(
      Zh, Ch, zsqn, csqn, prob_out, indices);

  // EMA statistics and codebook update
  stats_kernel<<<Nn, 256, 0, stream>>>(zn, indices, n_total, enc_sum);
  Nnew_kernel<<<Kk / 256, 256, 0, stream>>>(N_buf, n_total, Nnew_out);
  nsum_kernel<<<1, 256, 0, stream>>>(Nnew_out, nsum);
  embnew_kernel<<<Kk, 256, 0, stream>>>(z_avg, enc_sum, Nnew_out, nsum,
                                        zavgnew_out, embnew_out, cbsum_out);

  // emb gather + commitment loss, finalize, q transpose
  gather_loss_kernel<<<Nn, 256, 0, stream>>>(zn, indices, embbuf, loss_acc);
  finalize_kernel<<<1, 1, 0, stream>>>(loss_acc, commit_out, loss_out);
  qtranspose_kernel<<<dim3(HW / 32, Dd / 32, Bb), dim3(32, 32), 0, stream>>>(embbuf, q_out);
}
